// Head_10393820856593
// MI455X (gfx1250) — matrix operand order repaired and checked by reading, not compiled, not run on hardware
//
#include <hip/hip_runtime.h>

// ---------------------------------------------------------------------------
// Types for CDNA5 WMMA (wave32): v16 bf16 A/B fragments, v8 f32 C/D.
// ---------------------------------------------------------------------------
typedef __attribute__((ext_vector_type(8)))  float    v8f;
typedef __attribute__((ext_vector_type(4)))  float    v4f;
typedef __attribute__((ext_vector_type(8)))  __bf16   v8bf;
typedef __attribute__((ext_vector_type(16))) __bf16   v16bf;
typedef __attribute__((ext_vector_type(4)))  unsigned uint4v;
typedef __attribute__((ext_vector_type(8)))  int      int8v;
typedef __attribute__((ext_vector_type(4)))  int      int4v;

union BF16x16 { v16bf v; v8bf h[2]; };

__device__ __forceinline__ __bf16 f2bf(float f) {
  unsigned u = __builtin_bit_cast(unsigned, f);
  u += 0x7FFFu + ((u >> 16) & 1u);           // round-to-nearest-even
  unsigned short s = (unsigned short)(u >> 16);
  return __builtin_bit_cast(__bf16, s);
}

__device__ __forceinline__ v8f vzero8() {
  v8f z;
#pragma unroll
  for (int i = 0; i < 8; ++i) z[i] = 0.0f;
  return z;
}

// CDNA5 async global->LDS (ASYNCcnt path). VDST = per-lane LDS byte offset,
// VADDR = 64-bit global address, GV mode (saddr = off).
__device__ __forceinline__ void async_ld_b128(unsigned lds_off, const void* g) {
  asm volatile("global_load_async_to_lds_b128 %0, %1, off"
               :: "v"(lds_off), "v"(g) : "memory");
}
__device__ __forceinline__ void wait_async0() {
  asm volatile("s_wait_asynccnt 0x0" ::: "memory");
}
__device__ __forceinline__ unsigned lds_off(const void* p) {
  // addrspace(3) lives in the low 32 bits of the generic address
  return (unsigned)(unsigned long long)p;
}

// ---------------------------------------------------------------------------
// Tensor Data Mover: one descriptor-driven DMA for a 2D bf16 tile
// rows x 64 (row stride 64 elements), global -> LDS. Issued by one wave;
// completion tracked with TENSORcnt. D# layout per CDNA5 ISA 8.3/8.4.
// ---------------------------------------------------------------------------
__device__ __forceinline__ void tdm_load_tile64x64(unsigned lds_addr, const void* g) {
  const unsigned long long ga = (unsigned long long)g;
  uint4v g0;
  g0[0] = 1u;                                         // count=1 (valid), user D#
  g0[1] = lds_addr;                                   // lds_addr[31:0]
  g0[2] = (unsigned)ga;                               // global_addr[31:0]
  g0[3] = (unsigned)((ga >> 32) & 0x01FFFFFFu)        // global_addr[56:32]
        | (2u << 30);                                 // type=2 ("image")
  int8v g1;
  g1[0] = 0x10000;          // wg_mask=0 | data_size=1 (2 bytes) | no barrier/pad
  g1[1] = 64 << 16;         // tensor_dim0[15:0]=64 in [47:32..]: dim0 lo16
  g1[2] = 64 << 16;         // tensor_dim0 hi16=0 | tensor_dim1 lo16=64
  g1[3] = 64 << 16;         // tensor_dim1 hi16=0 | tile_dim0=64
  g1[4] = 64;               // tile_dim1=64 | tile_dim2=0
  g1[5] = 64;               // tensor_dim0_stride[31:0] = 64 elements
  g1[6] = 0;                // stride hi | tensor_dim1_stride lo (unused, 2D)
  g1[7] = 0;
  const int4v gz = { 0, 0, 0, 0 };
#if __clang_major__ >= 23
  const int8v gz8 = { 0, 0, 0, 0, 0, 0, 0, 0 };
  __builtin_amdgcn_tensor_load_to_lds(g0, g1, gz, gz, gz8, 0);
#else
  __builtin_amdgcn_tensor_load_to_lds(g0, g1, gz, gz, 0);
#endif
}
__device__ __forceinline__ void wait_tensor0() {
  __builtin_amdgcn_s_wait_tensorcnt(0);
}

// Dims fixed by the reference
#define EMBED 1024
#define HEAD  64
#define SEQ   2048
#define NTOK  (8 * SEQ)   // B*T = 16384

// ---------------------------------------------------------------------------
// Prep 1: x fp32 -> bf16, streaming. 8 elements / thread.
// ---------------------------------------------------------------------------
__global__ __launch_bounds__(256) void conv_x_kernel(
    const float* __restrict__ x, __bf16* __restrict__ Xbf)
{
  const size_t i = ((size_t)blockIdx.x * 256 + threadIdx.x) * 8;
  v4f f0 = *(const v4f*)(x + i);
  v4f f1 = *(const v4f*)(x + i + 4);
  v8bf o;
#pragma unroll
  for (int j = 0; j < 4; ++j) { o[j] = f2bf(f0[j]); o[4 + j] = f2bf(f1[j]); }
  *(v8bf*)(Xbf + i) = o;
}

// ---------------------------------------------------------------------------
// Prep 2: Wq/Wk/Wv fp32 [c][h] -> bf16 transposed Wt[m][h][c].
// ---------------------------------------------------------------------------
__global__ __launch_bounds__(256) void conv_w_kernel(
    const float* __restrict__ Wq, const float* __restrict__ Wk,
    const float* __restrict__ Wv, __bf16* __restrict__ Wt)
{
  const float* W[3] = { Wq, Wk, Wv };
  const int idx = blockIdx.x * 256 + threadIdx.x;   // over 3*1024*64
  const int m = idx >> 16;
  const int r = idx & 0xFFFF;
  const int c = r >> 6;
  const int h = r & 63;
  Wt[(size_t)m * (HEAD * EMBED) + h * EMBED + c] = f2bf(W[m][(size_t)c * HEAD + h]);
}

// ---------------------------------------------------------------------------
// Kernel 1: Q/K/V projection, double-buffered async weight staging.
// 128 rows/block, 8 waves x 16 rows. A-fragments straight from global bf16 x;
// weight chunk (12 KB) lands in LDS via async b128 while the previous chunk's
// 12 wmma execute.
// ---------------------------------------------------------------------------
__global__ __launch_bounds__(256) void qkv_proj_kernel(
    const __bf16* __restrict__ Xbf,
    const __bf16* __restrict__ Wt,
    __bf16* __restrict__ Qo,
    __bf16* __restrict__ Ko,
    __bf16* __restrict__ Vo)
{
  __shared__ __bf16 sWT[2][3 * 64 * 32];   // 2 x 12 KB, [m][h][c-local]

  const int tid  = threadIdx.x;
  const int lane = tid & 31;
  const int wave = tid >> 5;
  const int rowbase = blockIdx.x * 128;
  const int l15 = lane & 15;
  const int ch  = (lane & 16) ? 8 : 0;   // A-frag K-half selector
  const int cb  = (lane & 16) ? 16 : 0;  // B-frag K-half selector

  // per-thread fixed staging coordinates (3 b128 transfers / thread / chunk)
  int dsto[3];
  const __bf16* srcp[3];
#pragma unroll
  for (int it = 0; it < 3; ++it) {
    int e    = tid + it * 256;           // 0..767
    int mm   = e >> 8;
    int rr   = e & 255;
    int hh   = rr >> 2;
    int part = rr & 3;
    dsto[it] = mm * (64 * 32) + hh * 32 + part * 8;
    srcp[it] = Wt + (size_t)mm * (HEAD * EMBED) + hh * EMBED + part * 8;
  }

  v8f acc[3][4];
#pragma unroll
  for (int m = 0; m < 3; ++m)
#pragma unroll
    for (int n = 0; n < 4; ++n) acc[m][n] = vzero8();

  const __bf16* xr = Xbf + (size_t)(rowbase + wave * 16 + l15) * EMBED;

  // prologue: issue chunk 0 into buffer 0
#pragma unroll
  for (int it = 0; it < 3; ++it)
    async_ld_b128(lds_off(&sWT[0][dsto[it]]), srcp[it]);

  for (int kc = 0; kc < EMBED; kc += 32) {
    const int buf = (kc >> 5) & 1;
    wait_async0();                       // this chunk has landed (this wave)
    __syncthreads();                     // ...for all waves; prev readers done

    if (kc + 32 < EMBED) {               // issue next chunk into other buffer
#pragma unroll
      for (int it = 0; it < 3; ++it)
        async_ld_b128(lds_off(&sWT[buf ^ 1][dsto[it]]), srcp[it] + kc + 32);
    }

    // ---- A fragment: direct global bf16 loads, zero conversion ----
    BF16x16 a;
    a.h[0] = *(const v8bf*)(xr + kc + ch);
    a.h[1] = *(const v8bf*)(xr + kc + 16 + ch);

#pragma unroll
    for (int m = 0; m < 3; ++m) {
#pragma unroll
      for (int n = 0; n < 4; ++n) {
        BF16x16 bf;
        const __bf16* wr = &sWT[buf][m * (64 * 32) + (n * 16 + l15) * 32];
        bf.h[0] = *(const v8bf*)(wr + ch);
        bf.h[1] = *(const v8bf*)(wr + 16 + ch);
        acc[m][n] = __builtin_amdgcn_wmma_f32_16x16x32_bf16(
            false, a.v, false, bf.v, (short)0, acc[m][n], false, false);
      }
    }
  }

  // ---- store bf16 Q/K/V (C/D layout: VGPR r -> rows r / r+8) ----
  __bf16* Out[3] = { Qo, Ko, Vo };
  const int hi8 = (lane & 16) ? 8 : 0;
#pragma unroll
  for (int m = 0; m < 3; ++m)
#pragma unroll
    for (int n = 0; n < 4; ++n)
#pragma unroll
      for (int r = 0; r < 8; ++r) {
        int row = rowbase + wave * 16 + r + hi8;
        Out[m][(size_t)row * HEAD + n * 16 + l15] = f2bf(acc[m][n][r]);
      }
}

// ---------------------------------------------------------------------------
// Kernel 2: causal flash attention, double-buffered 64-key tiles.
// K tiles arrive via the Tensor Data Mover (one D# DMA per tile, issued by
// wave 0 one tile ahead, gated by s_wait_tensorcnt). V is fetched to
// registers one tile ahead and transposed into LDS at the top of the
// consuming iteration. 16 wmma per tile + exp2-domain online softmax.
// ---------------------------------------------------------------------------
__global__ __launch_bounds__(256) void attn_kernel(
    const __bf16* __restrict__ Q,
    const __bf16* __restrict__ K,
    const __bf16* __restrict__ V,
    float* __restrict__ out)
{
  __shared__ __bf16 sK [2][64 * 64];     // 2 x 8 KB, [key][c]   (TDM dest)
  __shared__ __bf16 sVt[2][64 * 64];     // 2 x 8 KB, [h][key]
  __shared__ __bf16 sP [8 * 16 * 64];    // 16 KB, per-wave P scratch

  const int tid  = threadIdx.x;
  const int lane = tid & 31;
  const int wave = tid >> 5;
  const int l15  = lane & 15;
  const int b    = blockIdx.y;
  const int qtb  = blockIdx.x;                  // 16 query tiles of 128
  const int qbase = qtb * 128 + wave * 16;      // within T
  const size_t tokb = (size_t)b * SEQ;

  const int ch  = (lane & 16) ? 8 : 0;          // A-frag K-half
  const int cb  = (lane & 16) ? 16 : 0;         // B-frag K-half
  const int hi8 = (lane & 16) ? 8 : 0;          // C/D row offset

  // V staging coordinates: two 16B slices per thread per tile
  const int e0 = tid * 8, e1 = tid * 8 + 2048;
  const int key0 = e0 >> 6, c0 = e0 & 63;
  const int key1 = e1 >> 6, c1 = e1 & 63;

  // Q fragments live in registers for the whole key loop (K-dim 64 = 2 frags)
  const __bf16* qr = Q + (tokb + qbase + l15) * HEAD;
  BF16x16 aq0, aq1;
  aq0.h[0] = *(const v8bf*)(qr + ch);
  aq0.h[1] = *(const v8bf*)(qr + 16 + ch);
  aq1.h[0] = *(const v8bf*)(qr + 32 + ch);
  aq1.h[1] = *(const v8bf*)(qr + 48 + ch);

  v8f O[4];
#pragma unroll
  for (int t = 0; t < 4; ++t) O[t] = vzero8();
  float m[8], l[8];
#pragma unroll
  for (int r = 0; r < 8; ++r) { m[r] = -1e30f; l[r] = 0.0f; }

  const int   qmax  = qbase + 15;
  const int   nkt   = (qtb + 1) * 2;            // 64-key tiles to cover block
  const float scale = 0.18033688f;              // log2(e) / sqrt(64)

  const __bf16* Kb = K + tokb * HEAD;
  const __bf16* Vb = V + tokb * HEAD;

  // prologue: tile 0 -> buffer 0 (K via TDM from wave 0, V to registers)
  if (wave == 0) tdm_load_tile64x64(lds_off(&sK[0][0]), Kb);
  v8bf vv0 = *(const v8bf*)(Vb + e0);
  v8bf vv1 = *(const v8bf*)(Vb + e1);

  for (int kt = 0; kt < nkt; ++kt) {
    const int kbase = kt * 64;
    const int buf   = kt & 1;

    // transpose this tile's V (fetched last iteration) into LDS
#pragma unroll
    for (int j = 0; j < 8; ++j) {
      sVt[buf][(c0 + j) * 64 + key0] = vv0[j];
      sVt[buf][(c1 + j) * 64 + key1] = vv1[j];
    }
    if (wave == 0) wait_tensor0();              // this tile's K DMA done
    __syncthreads();                            // visible to all waves

    if (kt + 1 < nkt) {                         // stage tile kt+1 into other buffer
      const __bf16* kg = Kb + (size_t)(kbase + 64) * HEAD;
      const __bf16* vg = Vb + (size_t)(kbase + 64) * HEAD;
      if (wave == 0) tdm_load_tile64x64(lds_off(&sK[buf ^ 1][0]), kg);
      vv0 = *(const v8bf*)(vg + e0);
      vv1 = *(const v8bf*)(vg + e1);
      if (kt + 2 < nkt) {                       // L2 prefetch two tiles ahead
        __builtin_prefetch(kg + 64 * HEAD + e0, 0, 0);
        __builtin_prefetch(vg + 64 * HEAD + e0, 0, 0);
      }
    }

    if (kbase > qmax) continue;                 // wave-uniform causal skip

    // ---- S = Q K^T for four 16-key subtiles (8 wmma) ----
    v8f S[4];
#pragma unroll
    for (int s = 0; s < 4; ++s) {
      const __bf16* kr = &sK[buf][(s * 16 + l15) * 64];
      BF16x16 b0, b1;
      b0.h[0] = *(const v8bf*)(kr + ch);
      b0.h[1] = *(const v8bf*)(kr + 16 + ch);
      b1.h[0] = *(const v8bf*)(kr + 32 + ch);
      b1.h[1] = *(const v8bf*)(kr + 48 + ch);
      v8f a = vzero8();
      a = __builtin_amdgcn_wmma_f32_16x16x32_bf16(
          false, aq1.v, false, b1.v, (short)0, a, false, false);
      a = __builtin_amdgcn_wmma_f32_16x16x32_bf16(
          false, aq0.v, false, b0.v, (short)0, a, false, false);
      S[s] = a;
    }

    // ---- online softmax across all 64 columns at once ----
#pragma unroll
    for (int r = 0; r < 8; ++r) {
      const int qrow = qbase + r + hi8;
      float v[4];
      bool  msk[4];
#pragma unroll
      for (int s = 0; s < 4; ++s) {
        msk[s] = (kbase + s * 16 + l15) > qrow;
        v[s]   = msk[s] ? -1e30f : S[s][r] * scale;
      }
      float rm = fmaxf(fmaxf(v[0], v[1]), fmaxf(v[2], v[3]));
      rm = fmaxf(rm, __shfl_xor(rm, 1));
      rm = fmaxf(rm, __shfl_xor(rm, 2));
      rm = fmaxf(rm, __shfl_xor(rm, 4));
      rm = fmaxf(rm, __shfl_xor(rm, 8));
      const float mn    = fmaxf(m[r], rm);
      const float alpha = exp2f(m[r] - mn);
      float p[4], rs = 0.0f;
#pragma unroll
      for (int s = 0; s < 4; ++s) {
        p[s] = msk[s] ? 0.0f : exp2f(v[s] - mn);
        rs  += p[s];
      }
      rs += __shfl_xor(rs, 1);
      rs += __shfl_xor(rs, 2);
      rs += __shfl_xor(rs, 4);
      rs += __shfl_xor(rs, 8);
      l[r] = l[r] * alpha + rs;
      m[r] = mn;
#pragma unroll
      for (int t = 0; t < 4; ++t) O[t][r] *= alpha;
      __bf16* pw = &sP[wave * 1024 + (r + hi8) * 64];
#pragma unroll
      for (int s = 0; s < 4; ++s) pw[s * 16 + l15] = f2bf(p[s]);
    }

    // order per-wave LDS writes before the A-layout re-read
    asm volatile("s_wait_dscnt 0x0" ::: "memory");

    // ---- O += P V (two A-frags over 64 keys, 8 wmma) ----
    BF16x16 ap0, ap1;
    const __bf16* pr = &sP[wave * 1024 + l15 * 64];
    ap0.h[0] = *(const v8bf*)(pr + ch);
    ap0.h[1] = *(const v8bf*)(pr + 16 + ch);
    ap1.h[0] = *(const v8bf*)(pr + 32 + ch);
    ap1.h[1] = *(const v8bf*)(pr + 48 + ch);
#pragma unroll
    for (int t = 0; t < 4; ++t) {
      const __bf16* vr = &sVt[buf][(t * 16 + l15) * 64];
      BF16x16 bv0, bv1;
      bv0.h[0] = *(const v8bf*)(vr + ch);
      bv0.h[1] = *(const v8bf*)(vr + 16 + ch);
      bv1.h[0] = *(const v8bf*)(vr + 32 + ch);
      bv1.h[1] = *(const v8bf*)(vr + 48 + ch);
      O[t] = __builtin_amdgcn_wmma_f32_16x16x32_bf16(
          false, ap0.v, false, bv0.v, (short)0, O[t], false, false);
      O[t] = __builtin_amdgcn_wmma_f32_16x16x32_bf16(
          false, ap1.v, false, bv1.v, (short)0, O[t], false, false);
    }
  }

  // ---- epilogue: O / l, fp32 store ----
#pragma unroll
  for (int r = 0; r < 8; ++r) {
    const float inv = 1.0f / l[r];
    const size_t row = tokb + qbase + r + hi8;
#pragma unroll
    for (int t = 0; t < 4; ++t)
      out[row * HEAD + t * 16 + l15] = O[t][r] * inv;
  }
}

// ---------------------------------------------------------------------------
// Launch: prep converts x/W to bf16 (Wt transposed), then WMMA-only passes.
// ws: Xbf 32MB | Wt 384KB | Q 2MB | K 2MB | V 2MB
// ---------------------------------------------------------------------------
extern "C" void kernel_launch(void* const* d_in, const int* in_sizes, int n_in,
                              void* d_out, int out_size, void* d_ws, size_t ws_size,
                              hipStream_t stream) {
  const float* x  = (const float*)d_in[0];
  const float* Wk = (const float*)d_in[1];
  const float* Wq = (const float*)d_in[2];
  const float* Wv = (const float*)d_in[3];

  __bf16* Xbf = (__bf16*)d_ws;
  __bf16* Wt  = Xbf + (size_t)NTOK * EMBED;
  __bf16* Q   = Wt + (size_t)3 * HEAD * EMBED;
  __bf16* K   = Q + (size_t)NTOK * HEAD;
  __bf16* V   = K + (size_t)NTOK * HEAD;

  conv_x_kernel<<<(NTOK * (EMBED / 8)) / 256, 256, 0, stream>>>(x, Xbf);
  conv_w_kernel<<<(3 * EMBED * HEAD) / 256, 256, 0, stream>>>(Wq, Wk, Wv, Wt);

  qkv_proj_kernel<<<NTOK / 128, 256, 0, stream>>>(Xbf, Wt, Q, K, V);

  dim3 grid(SEQ / 128, 8);
  attn_kernel<<<grid, 256, 0, stream>>>(Q, K, V, (float*)d_out);
}
